// HopeModel_18769007084155
// MI455X (gfx1250) — matrix operand order repaired and checked by reading, not compiled, not run on hardware
//
#include <hip/hip_runtime.h>

// ---------------------------------------------------------------------------
// HOPE forward for MI455X (gfx1250): bf16 WMMA GEMMs + wave32 scan/LN kernels
// Round 3: add Tensor Data Mover (TDM) staging of the weight tile when the
// toolchain exposes __builtin_amdgcn_tensor_load_to_lds (guarded fallback to
// the cooperative b128 staging path otherwise).
// ---------------------------------------------------------------------------

typedef __attribute__((ext_vector_type(16))) __bf16 v16bf;
typedef __attribute__((ext_vector_type(8)))  __bf16 v8bf;
typedef __attribute__((ext_vector_type(8)))  float  v8f;
typedef __attribute__((ext_vector_type(4)))  unsigned int v4u32;
typedef __attribute__((ext_vector_type(8)))  int    v8i32;
typedef __attribute__((ext_vector_type(4)))  int    v4i32;

#if defined(__has_builtin)
#if __has_builtin(__builtin_amdgcn_tensor_load_to_lds) && \
    __has_builtin(__builtin_amdgcn_s_wait_tensorcnt)
#define HOPE_HAVE_TDM 1
#endif
#endif

#define BB 2
#define SS 2048
#define DD 512
#define HH 2048
#define VV 32000
#define NRROWS (BB * SS)          // 4096 token rows
#define LN_EPS 1e-5f

// ---------------------------------------------------------------------------
// Tiled transpose + fp32->bf16 convert:  in (K x N fp32) -> out (N x K bf16)
// ---------------------------------------------------------------------------
__global__ __launch_bounds__(256)
void transpose_f32_to_bf16_kernel(const float* __restrict__ in,
                                  __bf16* __restrict__ out,
                                  int K, int N) {
  __shared__ float tile[32][33];
  const int k0 = blockIdx.y * 32;
  const int n0 = blockIdx.x * 32;
  for (int i = threadIdx.x; i < 32 * 32; i += 256) {
    int r = i >> 5, c = i & 31;                       // coalesced along n
    tile[r][c] = in[(size_t)(k0 + r) * N + (n0 + c)];
  }
  __syncthreads();
  if (threadIdx.x < 128) {
    int n = threadIdx.x >> 2;                         // 0..31
    int c = threadIdx.x & 3;                          // chunk of 8 k values
    v8bf v;
    #pragma unroll
    for (int e = 0; e < 8; ++e) v[e] = (__bf16)tile[c * 8 + e][n];
    *(v8bf*)(out + (size_t)(n0 + n) * K + k0 + c * 8) = v;
  }
}

// ---------------------------------------------------------------------------
// Embedding gather: emb[x[row]] -> bf16 activations, 8 elems (16B) per thread
// ---------------------------------------------------------------------------
__global__ __launch_bounds__(256)
void embed_gather_kernel(const int* __restrict__ x, const float* __restrict__ emb,
                         __bf16* __restrict__ out) {
  const int CH = DD / 8;                              // 64 chunks per row
  size_t total = (size_t)NRROWS * CH;
  size_t i = (size_t)blockIdx.x * blockDim.x + threadIdx.x;
  size_t stride = (size_t)gridDim.x * blockDim.x;
  for (; i < total; i += stride) {
    size_t row = i / CH;
    size_t ch  = i - row * CH;
    const float4* src = (const float4*)(emb + (size_t)x[row] * DD + ch * 8);
    float4 f0 = src[0], f1 = src[1];
    v8bf v;
    v[0] = (__bf16)f0.x; v[1] = (__bf16)f0.y; v[2] = (__bf16)f0.z; v[3] = (__bf16)f0.w;
    v[4] = (__bf16)f1.x; v[5] = (__bf16)f1.y; v[6] = (__bf16)f1.z; v[7] = (__bf16)f1.w;
    *(v8bf*)(out + row * DD + ch * 8) = v;
  }
}

// ---------------------------------------------------------------------------
// bf16 WMMA GEMM:  C(MxN) = A(MxK) @ W(KxN) + bias, optional ReLU.
// W is supplied TRANSPOSED (Wt: N x K) so both A and B WMMA fragments are
// contiguous K-runs in LDS -> ds_load_b128 only, no scalar packing.
// Block tile 64(M) x 128(N), 8 waves arranged 4x2; each wave owns a 16x64
// strip = 4 v_wmma_f32_16x16x32_bf16 accumulators reusing one A fragment.
// Wt tile staged by the Tensor Data Mover when available (D# descriptor with
// LDS padding reproducing the 96B row stride), else cooperative b128 copies.
// ---------------------------------------------------------------------------
template <bool RELU, bool WRITE_F32, bool WRITE_BF16>
__global__ __launch_bounds__(256)
void gemm_bf16_kernel(const __bf16* __restrict__ A,    // M x K row-major
                      const __bf16* __restrict__ Wt,   // N x K row-major
                      const float* __restrict__ bias,
                      float* __restrict__ Cf, __bf16* __restrict__ Cb,
                      int M, int N, int K) {
  const int tid   = threadIdx.x;
  const int wave  = tid >> 5;
  const int lane  = tid & 31;
  const int hsel  = lane >> 4;      // lane half (K-split select)
  const int l     = lane & 15;
  const int waveM = wave >> 1;      // 0..3
  const int waveN = wave & 1;       // 0..1
  const int rowBase = blockIdx.y * 64;
  const int colBase = blockIdx.x * 128;

  constexpr int LPAD = 48;          // LDS row stride in elems (96B, 16B-mult)
  __shared__ __align__(16) __bf16 As[64 * LPAD];    // 6 KB
  __shared__ __align__(16) __bf16 Bt[128 * LPAD];   // 12 KB

  v8f acc[4] = {{}, {}, {}, {}};

  for (int k0 = 0; k0 < K; k0 += 32) {
    // Stage A tile (64 rows x 32 K): 256 x 16B chunks, one per thread.
    {
      int r = tid >> 2, c = tid & 3;
      v8bf v = *(const v8bf*)(A + (size_t)(rowBase + r) * K + (k0 + c * 8));
      *(v8bf*)(As + r * LPAD + c * 8) = v;
    }

#ifdef HOPE_HAVE_TDM
    // Stage Wt tile (128 cols x 32 K) via the Tensor Data Mover. One wave
    // issues the DMA; D# pads the LDS destination by 8 DWORDs every 16 DWORDs
    // (64B rows -> 96B stride) to match the ds_load_b128 fragment addressing.
    if (wave == 0) {
      unsigned long long gaddr =
          (unsigned long long)(uintptr_t)(Wt + (size_t)colBase * K + k0);
      unsigned int ldsb = (unsigned int)(uintptr_t)(void*)Bt;
      v4u32 g0;
      g0[0] = 1u;                                         // count = 1 (valid)
      g0[1] = ldsb;                                       // lds_addr
      g0[2] = (unsigned int)gaddr;                        // global_addr[31:0]
      g0[3] = (unsigned int)((gaddr >> 32) & 0x01FFFFFFu) // global_addr[56:32]
              | (2u << 30);                               // type = 2 (image)
      v8i32 g1;
      g1[0] = (1 << 16)        // data_size = 1 (2 bytes)
            | (1 << 20)        // pad_enable
            | (3 << 22)        // pad_interval: 16 DWORDs (64B)
            | (7 << 25);       // pad_amount:   8 DWORDs (32B)
      g1[1] = (K & 0xFFFF) << 16;                         // tensor_dim0[15:0]
      g1[2] = ((K >> 16) & 0xFFFF) | ((N & 0xFFFF) << 16);// dim0 hi | dim1 lo
      g1[3] = ((N >> 16) & 0xFFFF) | (32 << 16);          // dim1 hi | tile_dim0
      g1[4] = 128;                                        // tile_dim1 (cols)
      g1[5] = K;                                          // dim0_stride[31:0]
      g1[6] = 0;
      g1[7] = 0;
      v4i32 z4 = {0, 0, 0, 0};
#if __clang_major__ >= 23
      v8i32 z8 = {0, 0, 0, 0, 0, 0, 0, 0};
      __builtin_amdgcn_tensor_load_to_lds(g0, g1, z4, z4, z8, 0);
#else
      __builtin_amdgcn_tensor_load_to_lds(g0, g1, z4, z4, 0);
#endif
      __builtin_amdgcn_s_wait_tensorcnt(0);
    }
#else
    // Stage Wt tile (128 cols x 32 K): 512 chunks, two per thread.
    #pragma unroll
    for (int i = 0; i < 2; ++i) {
      int idx = tid + i * 256;
      int n = idx >> 2, c = idx & 3;
      v8bf v = *(const v8bf*)(Wt + (size_t)(colBase + n) * K + (k0 + c * 8));
      *(v8bf*)(Bt + n * LPAD + c * 8) = v;
    }
    if (k0 + 32 < K) {
      __builtin_prefetch(Wt + (size_t)(colBase + (tid >> 1)) * K + (k0 + 32), 0, 1);
    }
#endif
    // Prefetch next A tile (global_prefetch_b8).
    if (k0 + 32 < K) {
      __builtin_prefetch(A + (size_t)(rowBase + (tid >> 2)) * K + (k0 + 32), 0, 1);
    }
    __syncthreads();

    // A fragment (16x32 bf16): lane half selects K {0..7,16..23}/{8..15,24..31},
    // both halves contiguous 16B runs in the row-major LDS tile.
    const __bf16* arow = As + (waveM * 16 + l) * LPAD;
    v8bf alo = *(const v8bf*)(arow + hsel * 8);
    v8bf ahi = *(const v8bf*)(arow + 16 + hsel * 8);
    v16bf a = __builtin_shufflevector(alo, ahi, 0, 1, 2, 3, 4, 5, 6, 7,
                                      8, 9, 10, 11, 12, 13, 14, 15);

    #pragma unroll
    for (int j = 0; j < 4; ++j) {
      // B fragment (32x16): column n = strip col, K {0..7,16..23}/{8..15,24..31} by lane half.
      const __bf16* brow = Bt + (waveN * 64 + j * 16 + l) * LPAD + hsel * 8;
      v8bf blo = *(const v8bf*)(brow);
      v8bf bhi = *(const v8bf*)(brow + 16);
      v16bf b = __builtin_shufflevector(blo, bhi, 0, 1, 2, 3, 4, 5, 6, 7,
                                        8, 9, 10, 11, 12, 13, 14, 15);
      acc[j] = __builtin_amdgcn_wmma_f32_16x16x32_bf16(
          false, a, false, b, (short)0, acc[j], false, false);
    }
    __syncthreads();
  }

  // Epilogue: C VGPR r -> (row = r + hsel*8, col = l) inside each 16x16 tile.
  #pragma unroll
  for (int j = 0; j < 4; ++j) {
    int col = colBase + waveN * 64 + j * 16 + l;
    float bcol = bias[col];
    #pragma unroll
    for (int r = 0; r < 8; ++r) {
      int row = rowBase + waveM * 16 + hsel * 8 + r;
      float v = acc[j][r] + bcol;
      if constexpr (RELU) v = fmaxf(v, 0.0f);
      size_t idx = (size_t)row * N + col;
      if constexpr (WRITE_F32)  Cf[idx] = v;
      if constexpr (WRITE_BF16) Cb[idx] = (__bf16)v;
    }
  }
}

// ---------------------------------------------------------------------------
// Wave-per-row dot(h_row, w) + b -> sigmoid  (eta / alpha second layer, N=1)
// ---------------------------------------------------------------------------
__global__ __launch_bounds__(256)
void rowdot_sigmoid_kernel(const __bf16* __restrict__ h, const float* __restrict__ w,
                           const float* __restrict__ b, float* __restrict__ out,
                           int rows, int K) {
  int wv = blockIdx.x * 8 + (threadIdx.x >> 5);
  int lane = threadIdx.x & 31;
  if (wv >= rows) return;
  float s = 0.0f;
  for (int c = lane; c < K; c += 32) s += (float)h[(size_t)wv * K + c] * w[c];
  #pragma unroll
  for (int off = 16; off > 0; off >>= 1) s += __shfl_xor(s, off, 32);
  if (lane == 0) out[wv] = 1.0f / (1.0f + expf(-(s + b[0])));
}

// ---------------------------------------------------------------------------
// decay[b,t] = alpha * (1 - eta * sum_d k^2)   (wave per row)
// ---------------------------------------------------------------------------
__global__ __launch_bounds__(256)
void scan_prep_kernel(const float* __restrict__ k, const float* __restrict__ eta,
                      const float* __restrict__ alpha, float* __restrict__ decay,
                      int rows) {
  int wv = blockIdx.x * 8 + (threadIdx.x >> 5);
  int lane = threadIdx.x & 31;
  if (wv >= rows) return;
  float s = 0.0f;
  for (int c = lane; c < DD; c += 32) {
    float kv = k[(size_t)wv * DD + c];
    s += kv * kv;
  }
  #pragma unroll
  for (int off = 16; off > 0; off >>= 1) s += __shfl_xor(s, off, 32);
  if (lane == 0) decay[wv] = alpha[wv] * (1.0f - eta[wv] * s);
}

// ---------------------------------------------------------------------------
// Delta-rule fast-weight scan: rows of M evolve independently.
// One wave per (batch, row i); 512 fp32 state = 16 regs/lane across S steps.
//   M_i *= decay;  err = M_i . k - vhk_i;  M_i -= eta * err * k
// ---------------------------------------------------------------------------
__global__ __launch_bounds__(256)
void delta_scan_kernel(const float* __restrict__ k, const float* __restrict__ vhk,
                       const float* __restrict__ eta, const float* __restrict__ decay,
                       float* __restrict__ Mout) {
  const int PER = DD / 32;                         // 16 elems per lane
  int wv = blockIdx.x * 8 + (threadIdx.x >> 5);
  int lane = threadIdx.x & 31;
  if (wv >= BB * DD) return;
  int b = wv / DD;
  int i = wv - b * DD;

  float Mrow[16];
  #pragma unroll
  for (int j = 0; j < 16; ++j) Mrow[j] = 0.0f;

  for (int t = 0; t < SS; ++t) {
    const size_t rt = (size_t)b * SS + t;
    const float4* kt4 = (const float4*)(k + rt * DD + lane * PER);
    float4 kq[4];
    #pragma unroll
    for (int q = 0; q < 4; ++q) kq[q] = kt4[q];
    const float* kv = (const float*)kq;
    float part = 0.0f;
    #pragma unroll
    for (int j = 0; j < 16; ++j) part += Mrow[j] * kv[j];
    #pragma unroll
    for (int off = 16; off > 0; off >>= 1) part += __shfl_xor(part, off, 32);
    float dt = decay[rt];
    float et = eta[rt];
    float err = dt * part - vhk[rt * DD + i];      // dot with decayed M
    float coef = et * err;
    #pragma unroll
    for (int j = 0; j < 16; ++j) Mrow[j] = Mrow[j] * dt - coef * kv[j];
  }
  float4* dst = (float4*)(Mout + ((size_t)b * DD + i) * DD + lane * PER);
  #pragma unroll
  for (int q = 0; q < 4; ++q) {
    float4 o;
    o.x = Mrow[q * 4 + 0]; o.y = Mrow[q * 4 + 1];
    o.z = Mrow[q * 4 + 2]; o.w = Mrow[q * 4 + 3];
    dst[q] = o;
  }
}

// ---------------------------------------------------------------------------
// CMS residual LayerNorm: cur += LN(y)*gamma + beta   (wave per row, D=512)
// ---------------------------------------------------------------------------
__global__ __launch_bounds__(256)
void ln_residual_kernel(const float* __restrict__ y, const float* __restrict__ gamma,
                        const float* __restrict__ beta, float* __restrict__ cur,
                        __bf16* __restrict__ curb, int rows) {
  const int PER = DD / 32;
  int wv = blockIdx.x * 8 + (threadIdx.x >> 5);
  int lane = threadIdx.x & 31;
  if (wv >= rows) return;

  float yv[16];
  float s = 0.0f, s2 = 0.0f;
  #pragma unroll
  for (int j = 0; j < 16; ++j) {
    yv[j] = y[(size_t)wv * DD + lane * PER + j];
    s  += yv[j];
    s2 += yv[j] * yv[j];
  }
  #pragma unroll
  for (int off = 16; off > 0; off >>= 1) {
    s  += __shfl_xor(s, off, 32);
    s2 += __shfl_xor(s2, off, 32);
  }
  float mu  = s * (1.0f / DD);
  float var = s2 * (1.0f / DD) - mu * mu;
  float rstd = rsqrtf(var + LN_EPS);
  #pragma unroll
  for (int j = 0; j < 16; ++j) {
    int d = lane * PER + j;
    size_t idx = (size_t)wv * DD + d;
    float o = cur[idx] + (yv[j] - mu) * rstd * gamma[d] + beta[d];
    cur[idx]  = o;
    curb[idx] = (__bf16)o;
  }
}

// ---------------------------------------------------------------------------
// Host-side orchestration
// ---------------------------------------------------------------------------
enum {
  IN_X = 0, IN_STEP = 1, IN_EMB = 2,
  MEMK_W1 = 3,  MEMK_B1, MEMK_W2, MEMK_B2,
  MEMV_W1 = 7,  MEMV_B1, MEMV_W2, MEMV_B2,
  MEMQ_W1 = 11, MEMQ_B1, MEMQ_W2, MEMQ_B2,
  MEMM_W1 = 15, MEMM_B1, MEMM_W2, MEMM_B2,
  ETA_W1 = 19,  ETA_B1, ETA_W2, ETA_B2,
  ALP_W1 = 23,  ALP_B1, ALP_W2, ALP_B2,
  VGK_W = 27,   VGK_B,
  VGV_W = 29,   VGV_B,
  CMS0_W1 = 31, CMS0_B1, CMS0_W2, CMS0_B2, CMS0_G, CMS0_BT,
  CMS1_W1 = 37, CMS1_B1, CMS1_W2, CMS1_B2, CMS1_G, CMS1_BT,
  CMS2_W1 = 43, CMS2_B1, CMS2_W2, CMS2_B2, CMS2_G, CMS2_BT,
  OUT_W = 49,   OUT_B = 50
};

extern "C" void kernel_launch(void* const* d_in, const int* in_sizes, int n_in,
                              void* d_out, int out_size, void* d_ws, size_t ws_size,
                              hipStream_t stream) {
  (void)in_sizes; (void)n_in; (void)out_size; (void)ws_size;

  const int*   x_ids = (const int*)d_in[IN_X];
  const float* emb   = (const float*)d_in[IN_EMB];
  auto F = [&](int i) { return (const float*)d_in[i]; };

  // ---- workspace bump allocator (256B aligned) ----
  size_t off = 0;
  auto alloc = [&](size_t bytes) -> void* {
    off = (off + 255) & ~(size_t)255;
    void* p = (char*)d_ws + off;
    off += bytes;
    return p;
  };

  const size_t DH   = (size_t)DD * HH;
  const size_t DDsq = (size_t)DD * DD;
  const size_t DV   = (size_t)DD * VV;

  // bf16 transposed weight copies (N x K)
  __bf16* wk1 = (__bf16*)alloc(DH * 2);   __bf16* wk2 = (__bf16*)alloc(DH * 2);
  __bf16* wv1 = (__bf16*)alloc(DH * 2);   __bf16* wv2 = (__bf16*)alloc(DH * 2);
  __bf16* wq1 = (__bf16*)alloc(DH * 2);   __bf16* wq2 = (__bf16*)alloc(DH * 2);
  __bf16* wm1 = (__bf16*)alloc(DH * 2);   __bf16* wm2 = (__bf16*)alloc(DH * 2);
  __bf16* we1 = (__bf16*)alloc(DH * 2);
  __bf16* wa1 = (__bf16*)alloc(DH * 2);
  __bf16* wvg = (__bf16*)alloc(DDsq * 2);
  __bf16* wc1[3]; __bf16* wc2[3];
  for (int i = 0; i < 3; ++i) { wc1[i] = (__bf16*)alloc(DH * 2); wc2[i] = (__bf16*)alloc(DH * 2); }
  __bf16* wout = (__bf16*)alloc(DV * 2);

  // activations
  __bf16* emb_b = (__bf16*)alloc((size_t)NRROWS * DD * 2);
  __bf16* h_b   = (__bf16*)alloc((size_t)NRROWS * HH * 2);
  __bf16* v_b   = (__bf16*)alloc((size_t)NRROWS * DD * 2);
  __bf16* q_b   = (__bf16*)alloc((size_t)NRROWS * DD * 2);
  __bf16* cur_b = (__bf16*)alloc((size_t)NRROWS * DD * 2);
  float* k_f    = (float*)alloc((size_t)NRROWS * DD * 4);
  float* vhk_f  = (float*)alloc((size_t)NRROWS * DD * 4);
  float* cur_f  = (float*)alloc((size_t)NRROWS * DD * 4);
  float* y_f    = (float*)alloc((size_t)NRROWS * DD * 4);
  float* eta_f  = (float*)alloc((size_t)NRROWS * 4);
  float* alp_f  = (float*)alloc((size_t)NRROWS * 4);
  float* dec_f  = (float*)alloc((size_t)NRROWS * 4);

  dim3 blk(256);

  // ---- 1. weight transpose+convert (K x N fp32 -> N x K bf16) ----
  auto tcvt = [&](int idx, __bf16* dst, int K, int N) {
    dim3 g(N / 32, K / 32);
    transpose_f32_to_bf16_kernel<<<g, blk, 0, stream>>>(F(idx), dst, K, N);
  };
  tcvt(MEMK_W1, wk1, DD, HH);  tcvt(MEMK_W2, wk2, HH, DD);
  tcvt(MEMV_W1, wv1, DD, HH);  tcvt(MEMV_W2, wv2, HH, DD);
  tcvt(MEMQ_W1, wq1, DD, HH);  tcvt(MEMQ_W2, wq2, HH, DD);
  tcvt(MEMM_W1, wm1, DD, HH);  tcvt(MEMM_W2, wm2, HH, DD);
  tcvt(ETA_W1,  we1, DD, HH);  tcvt(ALP_W1,  wa1, DD, HH);
  tcvt(VGK_W,   wvg, DD, DD);
  tcvt(CMS0_W1, wc1[0], DD, HH); tcvt(CMS0_W2, wc2[0], HH, DD);
  tcvt(CMS1_W1, wc1[1], DD, HH); tcvt(CMS1_W2, wc2[1], HH, DD);
  tcvt(CMS2_W1, wc1[2], DD, HH); tcvt(CMS2_W2, wc2[2], HH, DD);
  tcvt(OUT_W,   wout, DD, VV);

  // ---- 2. embedding gather ----
  embed_gather_kernel<<<512, blk, 0, stream>>>(x_ids, emb, emb_b);

  dim3 gH(HH / 128, NRROWS / 64);    // N=2048 GEMMs
  dim3 gD(DD / 128, NRROWS / 64);    // N=512  GEMMs
  dim3 gV(VV / 128, NRROWS / 64);    // N=32000 output GEMM
  const int rowBlocks = NRROWS / 8;  // wave-per-row kernels

  // ---- 3. mem_k / mem_v / mem_q MLPs ----
  gemm_bf16_kernel<true,  false, true ><<<gH, blk, 0, stream>>>(emb_b, wk1, F(MEMK_B1), nullptr, h_b, NRROWS, HH, DD);
  gemm_bf16_kernel<false, true,  false><<<gD, blk, 0, stream>>>(h_b,   wk2, F(MEMK_B2), k_f, nullptr, NRROWS, DD, HH);
  gemm_bf16_kernel<true,  false, true ><<<gH, blk, 0, stream>>>(emb_b, wv1, F(MEMV_B1), nullptr, h_b, NRROWS, HH, DD);
  gemm_bf16_kernel<false, false, true ><<<gD, blk, 0, stream>>>(h_b,   wv2, F(MEMV_B2), nullptr, v_b, NRROWS, DD, HH);
  gemm_bf16_kernel<true,  false, true ><<<gH, blk, 0, stream>>>(emb_b, wq1, F(MEMQ_B1), nullptr, h_b, NRROWS, HH, DD);
  gemm_bf16_kernel<false, false, true ><<<gD, blk, 0, stream>>>(h_b,   wq2, F(MEMQ_B2), nullptr, q_b, NRROWS, DD, HH);

  // ---- 4. eta / alpha gates ----
  gemm_bf16_kernel<true, false, true><<<gH, blk, 0, stream>>>(emb_b, we1, F(ETA_B1), nullptr, h_b, NRROWS, HH, DD);
  rowdot_sigmoid_kernel<<<rowBlocks, blk, 0, stream>>>(h_b, F(ETA_W2), F(ETA_B2), eta_f, NRROWS, HH);
  gemm_bf16_kernel<true, false, true><<<gH, blk, 0, stream>>>(emb_b, wa1, F(ALP_B1), nullptr, h_b, NRROWS, HH, DD);
  rowdot_sigmoid_kernel<<<rowBlocks, blk, 0, stream>>>(h_b, F(ALP_W2), F(ALP_B2), alp_f, NRROWS, HH);

  // ---- 5. v_hat_k = v @ vg_k.W + b ----
  gemm_bf16_kernel<false, true, false><<<gD, blk, 0, stream>>>(v_b, wvg, F(VGK_B), vhk_f, nullptr, NRROWS, DD, DD);

  // ---- 6. delta-rule scan -> new_mem_k (tail of d_out) ----
  scan_prep_kernel<<<rowBlocks, blk, 0, stream>>>(k_f, eta_f, alp_f, dec_f, NRROWS);
  float* mem_out = (float*)d_out + (size_t)NRROWS * VV;
  delta_scan_kernel<<<(BB * DD) / 8, blk, 0, stream>>>(k_f, vhk_f, eta_f, dec_f, mem_out);

  // ---- 7. x_mod = MLP(mem_memory, q) ----
  gemm_bf16_kernel<true,  false, true><<<gH, blk, 0, stream>>>(q_b, wm1, F(MEMM_B1), nullptr, h_b, NRROWS, HH, DD);
  gemm_bf16_kernel<false, true,  true><<<gD, blk, 0, stream>>>(h_b, wm2, F(MEMM_B2), cur_f, cur_b, NRROWS, DD, HH);

  // ---- 8. CMS levels (global_step == 0 -> all frequencies active) ----
  const int cmsIdx[3][4] = {
    {CMS0_B1, CMS0_B2, CMS0_G, CMS0_BT},
    {CMS1_B1, CMS1_B2, CMS1_G, CMS1_BT},
    {CMS2_B1, CMS2_B2, CMS2_G, CMS2_BT},
  };
  __bf16* wcs1[3] = {wc1[0], wc1[1], wc1[2]};
  __bf16* wcs2[3] = {wc2[0], wc2[1], wc2[2]};
  for (int i = 0; i < 3; ++i) {
    gemm_bf16_kernel<true,  false, true ><<<gH, blk, 0, stream>>>(cur_b, wcs1[i], F(cmsIdx[i][0]), nullptr, h_b, NRROWS, HH, DD);
    gemm_bf16_kernel<false, true,  false><<<gD, blk, 0, stream>>>(h_b,   wcs2[i], F(cmsIdx[i][1]), y_f, nullptr, NRROWS, DD, HH);
    ln_residual_kernel<<<rowBlocks, blk, 0, stream>>>(y_f, F(cmsIdx[i][2]), F(cmsIdx[i][3]), cur_f, cur_b, NRROWS);
  }

  // ---- 9. logits = cur @ out.W + out.b -> d_out head ----
  gemm_bf16_kernel<false, true, false><<<gV, blk, 0, stream>>>(cur_b, wout, F(OUT_B), (float*)d_out, nullptr, NRROWS, VV, DD);
}
